// EdgeGNN_53008486367981
// MI455X (gfx1250) — matrix operand order repaired and checked by reading, not compiled, not run on hardware
//
#include <hip/hip_runtime.h>
#include <math.h>

// ---------------------------------------------------------------------------
// EdgeGNN for MI455X (gfx1250): f16 WMMA GEMMs, LDS-free.
//  - Weights pre-packed in WMMA B-fragment order (L0/L2 resident, 32B/lane loads)
//  - A fragments loaded directly from global (no cross-wave A reuse exists)
//  - gather (x[src]||x[dst]) fused into A loads; segment-sum fused into epilogue
// ---------------------------------------------------------------------------

typedef __attribute__((ext_vector_type(16))) _Float16 v16h;
typedef __attribute__((ext_vector_type(8)))  _Float16 v8h;
typedef __attribute__((ext_vector_type(8)))  float    v8f;

#define NN 100000   // nodes
#define NE 600000   // edges
#define NF 64       // input feat
#define NH 128      // hidden

// ---------------------------------------------------------------------------
// Utility kernels
// ---------------------------------------------------------------------------
__global__ void cvt_f32_to_f16(const float* __restrict__ in,
                               _Float16* __restrict__ out, long long n) {
    long long i = (long long)blockIdx.x * blockDim.x + threadIdx.x;
    long long stride = (long long)gridDim.x * blockDim.x;
    for (; i < n; i += stride) out[i] = (_Float16)in[i];
}

__global__ void zero_f32(float* __restrict__ p, long long n) {
    long long i = (long long)blockIdx.x * blockDim.x + threadIdx.x;
    long long stride = (long long)gridDim.x * blockDim.x;
    for (; i < n; i += stride) p[i] = 0.0f;
}

// Pack W[K,N] (f32) into WMMA B-fragment order (f16):
//   out[(((kcidx*(N/16) + nAll)*32 + lane)*16 + e] = W[kcidx*32 + e + 16*(lane>=16)][nAll*16 + (lane&15)]
// One thread per (kcidx, nAll, lane) -> writes 16 halves (32B contiguous).
__global__ void pack_B(const float* __restrict__ W, _Float16* __restrict__ out,
                       int K, int N) {
    int t = blockIdx.x * 256 + threadIdx.x;
    int nTiles = N >> 4;
    int total  = (K >> 5) * nTiles * 32;
    if (t >= total) return;
    int lane  = t & 31;
    int nAll  = (t >> 5) % nTiles;
    int kcidx = t / (32 * nTiles);
    int hi  = lane >> 4;
    int col = nAll * 16 + (lane & 15);
    _Float16* dst = out + (size_t)t * 16;
#pragma unroll
    for (int e = 0; e < 16; ++e)
        dst[e] = (_Float16)W[(size_t)(kcidx * 32 + e + hi * 16) * N + col];
}

// ---------------------------------------------------------------------------
// LDS-free tiled GEMM + bias + ReLU using v_wmma_f32_16x16x32_f16.
//   C[M,Nfull] = relu(A[M,K] @ B[K,Nfull] + bias)
// Block: 256 threads = 8 waves. Block tile 128(M) x 128(N) (grid.y spans Nfull).
// Each wave: 16 rows x 128 cols -> 8 f32 accumulators.
// GATHER:  A row e = concat(Anode[src[e]], Anode[dst[e]])  (K = 2*rowlen)
// SCATTER: epilogue atomically adds C row e into scat[src[e]][col] and
//          scat[dst[e]][128+col]  (segment-sum fused into GEMM).
// ---------------------------------------------------------------------------
template <bool GATHER, bool SCATTER>
__global__ __launch_bounds__(256)
void gemm_relu(const _Float16* __restrict__ A,
               const int* __restrict__ gsrc, const int* __restrict__ gdst,
               const _Float16* __restrict__ Bp,    // packed weights (fragment order)
               const float* __restrict__ bias,     // [Nfull]
               _Float16* __restrict__ C16,         // [M, Nfull]
               float* __restrict__ C32,            // optional f32 copy (z), may be null
               float* __restrict__ scat,           // [*, 256] f32 accumulator
               int M, int K, int Nfull) {
    const int tid     = threadIdx.x;
    const int rowBase = blockIdx.x * 128;
    const int colBase = blockIdx.y * 128;
    const int wave    = tid >> 5;        // 0..7 (wave32)
    const int lane    = tid & 31;
    const int hi      = (lane >> 4) & 1;
    const int l15     = lane & 15;
    const int nTiles  = Nfull >> 4;      // B-fragment tiles per K chunk

    v8f acc[8];
#pragma unroll
    for (int n = 0; n < 8; ++n)
#pragma unroll
        for (int r = 0; r < 8; ++r) acc[n][r] = 0.0f;

    // This lane's A row (clamped; guards only the stores/atomics below).
    int rowA = rowBase + wave * 16 + l15;
    if (rowA >= M) rowA = M - 1;

    const int halfK = K >> 1;            // gather mode: per-node feature width
    int nodeS = 0, nodeD = 0;
    if (GATHER) { nodeS = gsrc[rowA]; nodeD = gdst[rowA]; }

    for (int kc = 0; kc < K; kc += 32) {
        // ---- A fragment: two contiguous 16B chunks of this lane's row
        //      element e : k = e + hi*16
        const _Float16* ar;
        if (GATHER) {
            int side = (kc >= halfK) ? 1 : 0;
            int node = side ? nodeD : nodeS;
            ar = A + (size_t)node * halfK + (kc - side * halfK);
        } else {
            ar = A + (size_t)rowA * K + kc;
        }
        union { v16h v; v8h h[2]; } au;
        au.h[0] = *(const v8h*)(ar + hi * 16);
        au.h[1] = *(const v8h*)(ar + hi * 16 + 8);

        // ---- 8 WMMAs across the 128-wide N tile; B fragments are direct
        //      32B/lane loads from the packed (L2-resident) weight buffer.
        const _Float16* bbase =
            Bp + ((size_t)((kc >> 5) * nTiles + (colBase >> 4)) * 32 + lane) * 16;
#pragma unroll
        for (int n = 0; n < 8; ++n) {
            v16h b = *(const v16h*)(bbase + (size_t)n * 32 * 16);
            acc[n] = __builtin_amdgcn_wmma_f32_16x16x32_f16(
                false, au.v, false, b, (short)0, acc[n], false, false);
        }
    }

    // ---- epilogue: bias + relu, f16 store, optional f32 store, fused scatter
#pragma unroll
    for (int n = 0; n < 8; ++n) {
        int col  = colBase + n * 16 + l15;
        float bv = bias[col];
#pragma unroll
        for (int r = 0; r < 8; ++r) {
            int row = rowBase + wave * 16 + r + hi * 8;  // D layout: M = r + 8*hi
            float v = acc[n][r] + bv;
            v = v > 0.0f ? v : 0.0f;
            if (row < M) {
                C16[(size_t)row * Nfull + col] = (_Float16)v;
                if (C32) C32[(size_t)row * Nfull + col] = v;
                if (SCATTER) {
                    int s = gsrc[row], d = gdst[row];
                    atomicAdd(&scat[(size_t)s * 256 + col], v);        // x_in
                    atomicAdd(&scat[(size_t)d * 256 + 128 + col], v);  // x_out
                }
            }
        }
    }
}

// ---------------------------------------------------------------------------
// Decoder: preds[e] = sigmoid(dot(z[e,:], W_dec) + b_dec)
// ---------------------------------------------------------------------------
__global__ void decoder_kernel(const _Float16* __restrict__ z,
                               const float* __restrict__ w,
                               const float* __restrict__ b,
                               float* __restrict__ preds, int E) {
    int e = blockIdx.x * 256 + threadIdx.x;
    if (e >= E) return;
    const _Float16* zr = z + (size_t)e * NH;
    float s = b[0];
#pragma unroll 8
    for (int k = 0; k < NH; ++k) s += (float)zr[k] * w[k];
    preds[e] = 1.0f / (1.0f + expf(-s));
}

// ---------------------------------------------------------------------------
extern "C" void kernel_launch(void* const* d_in, const int* in_sizes, int n_in,
                              void* d_out, int out_size, void* d_ws, size_t ws_size,
                              hipStream_t stream) {
    const float* inputs = (const float*)d_in[0];
    const float* W_n11  = (const float*)d_in[1];  const float* b_n11 = (const float*)d_in[2];
    const float* W_n21  = (const float*)d_in[3];  const float* b_n21 = (const float*)d_in[4];
    const float* W_n22  = (const float*)d_in[5];  const float* b_n22 = (const float*)d_in[6];
    const float* W_e11  = (const float*)d_in[7];  const float* b_e11 = (const float*)d_in[8];
    const float* W_e12  = (const float*)d_in[9];  const float* b_e12 = (const float*)d_in[10];
    const float* W_dec  = (const float*)d_in[11]; const float* b_dec = (const float*)d_in[12];
    const int*   src    = (const int*)d_in[13];
    const int*   dst    = (const int*)d_in[14];
    float* out = (float*)d_out;          // [NE preds][NE*NH z]

    // ---- workspace carve-out (256B aligned slices)
    char* ws = (char*)d_ws;
    size_t off = 0;
    auto take = [&](size_t bytes) -> char* {
        char* p = ws + off;
        off = (off + bytes + 255) & ~(size_t)255;
        return p;
    };
    _Float16* wp_n11 = (_Float16*)take((size_t)NF * 256 * 2);   // packed weights
    _Float16* wp_n21 = (_Float16*)take((size_t)256 * NH * 2);
    _Float16* wp_n22 = (_Float16*)take((size_t)NH * NH * 2);
    _Float16* wp_e11 = (_Float16*)take((size_t)256 * NH * 2);
    _Float16* wp_e12 = (_Float16*)take((size_t)NH * NH * 2);
    _Float16* in16   = (_Float16*)take((size_t)NN * NF * 2);
    _Float16* X16    = (_Float16*)take((size_t)NN * 256 * 2);   // node state [N,256]
    _Float16* H1     = (_Float16*)take((size_t)NN * NH * 2);
    _Float16* H2     = (_Float16*)take((size_t)NN * NH * 2);
    _Float16* XE1    = (_Float16*)take((size_t)NE * NH * 2);
    _Float16* Z16    = (_Float16*)take((size_t)NE * NH * 2);
    float*    XACC   = (float*)   take((size_t)NN * 256 * 4);   // scatter accumulator

    // ---- pack weights into WMMA B-fragment order (fuses f32->f16 convert)
    auto pack = [&](const float* W, _Float16* outp, int K, int N) {
        int total = (K >> 5) * (N >> 4) * 32;
        pack_B<<<(total + 255) / 256, 256, 0, stream>>>(W, outp, K, N);
    };
    pack(W_n11, wp_n11, NF, 256);
    pack(W_n21, wp_n21, 256, NH);
    pack(W_n22, wp_n22, NH, NH);
    pack(W_e11, wp_e11, 256, NH);
    pack(W_e12, wp_e12, NH, NH);
    cvt_f32_to_f16<<<4096, 256, 0, stream>>>(inputs, in16, (long long)NN * NF);

    const dim3 blk(256);
    const dim3 gNode((NN + 127) / 128, 1);
    const dim3 gNode2((NN + 127) / 128, 2);   // Nout = 256 -> two 128-col tiles
    const dim3 gEdge((NE + 127) / 128, 1);

    // ---- encoder: x = relu(inputs @ W_n11 + b)   [N,64] -> [N,256]
    gemm_relu<false, false><<<gNode2, blk, 0, stream>>>(
        in16, nullptr, nullptr, wp_n11, b_n11, X16, nullptr, nullptr, NN, NF, 256);

    for (int it = 0; it < 3; ++it) {
        const bool last = (it == 2);
        if (!last)
            zero_f32<<<4096, 256, 0, stream>>>(XACC, (long long)NN * 256);

        // node MLP
        gemm_relu<false, false><<<gNode, blk, 0, stream>>>(
            X16, nullptr, nullptr, wp_n21, b_n21, H1, nullptr, nullptr, NN, 256, NH);
        gemm_relu<false, false><<<gNode, blk, 0, stream>>>(
            H1, nullptr, nullptr, wp_n22, b_n22, H2, nullptr, nullptr, NN, NH, NH);

        // edge MLP layer 1: gather concat(x[src], x[dst]) fused into A loads
        gemm_relu<true, false><<<gEdge, blk, 0, stream>>>(
            H2, src, dst, wp_e11, b_e11, XE1, nullptr, nullptr, NE, 256, NH);

        // edge MLP layer 2: segment-sum fused into epilogue (skipped last iter
        // where the scatter result is unused); last iter also writes z (f32)
        // directly into the output buffer.
        if (!last) {
            gemm_relu<false, true><<<gEdge, blk, 0, stream>>>(
                XE1, src, dst, wp_e12, b_e12, Z16, nullptr, XACC, NE, NH, NH);
            // x <- f16(concat(x_in, x_out)) for next iteration
            cvt_f32_to_f16<<<4096, 256, 0, stream>>>(XACC, X16, (long long)NN * 256);
        } else {
            gemm_relu<false, false><<<gEdge, blk, 0, stream>>>(
                XE1, nullptr, nullptr, wp_e12, b_e12, Z16, out + NE, nullptr, NE, NH, NH);
        }
    }

    // ---- decoder: preds = sigmoid(z @ W_dec + b_dec)
    decoder_kernel<<<(NE + 255) / 256, blk, 0, stream>>>(Z16, W_dec, b_dec, out, NE);
}
